// MultiHeadAttention_884763263242
// MI455X (gfx1250) — matrix operand order repaired and checked by reading, not compiled, not run on hardware
//
#include <hip/hip_runtime.h>
#include <hip/hip_bf16.h>

// ---------------------------------------------------------------------------
// MHA forward for B=4, S=2048, D=1024, H=16, HD=64 (causal, fp32 I/O)
// Matmuls via V_WMMA_F32_16X16X32_BF16; tile staging via TDM
// (TENSOR_LOAD_TO_LDS) where available.  CDNA5 / gfx1250, wave32.
// ---------------------------------------------------------------------------

typedef __attribute__((ext_vector_type(16))) __bf16 v16bf;
typedef __attribute__((ext_vector_type(8)))  float  v8f;
typedef __attribute__((ext_vector_type(4)))  unsigned int tdm_u32x4;
typedef __attribute__((ext_vector_type(8)))  int          tdm_i32x8;
typedef __attribute__((ext_vector_type(4)))  int          tdm_i32x4;

#define S_LEN   2048
#define NHEAD   16
#define HDIM    64
#define DMODEL  1024
#define MROWS   8192            // B * S
#define SCALE   0.125f          // 1/sqrt(64)
#define LOG2E   1.4426950408889634f
#define AW      4               // waves per attention block

#if __has_builtin(__builtin_amdgcn_tensor_load_to_lds)
#define USE_TDM 1
#else
#define USE_TDM 0
#endif

#if USE_TDM
// Issue a 2D TDM load: tile [tile_d1 rows x tile_d0 elems] of bf16 (2 bytes)
// from global (row stride = stride0 elems) into LDS at byte offset lds_off.
// D# layout per CDNA5 ISA.
__device__ __forceinline__ void tdm_load_2d_bf16(
    unsigned lds_off, const void* gaddr,
    unsigned tensor_d0, unsigned tensor_d1,
    unsigned tile_d0, unsigned tile_d1, unsigned stride0) {
  const unsigned long long ga = (unsigned long long)(size_t)gaddr;
  tdm_u32x4 g0;
  g0[0] = 1u;                                      // count=1, user descriptor
  g0[1] = lds_off;                                 // lds_addr (bytes)
  g0[2] = (unsigned)(ga & 0xffffffffu);            // global_addr[31:0]
  g0[3] = (unsigned)((ga >> 32) & 0x01ffffffu)     // global_addr[56:32]
          | (2u << 30);                            // type = 2 ("image")
  tdm_i32x8 g1;
  g1[0] = (int)(1u << 16);                         // data_size=1 (2 bytes)
  g1[1] = (int)(tensor_d0 << 16);                  // tensor_dim0[15:0] @ [31:16]
  g1[2] = (int)((tensor_d0 >> 16) | (tensor_d1 << 16));
  g1[3] = (int)((tensor_d1 >> 16) | (tile_d0 << 16)); // tile_dim0 @ [127:112]
  g1[4] = (int)(tile_d1 & 0xffffu);                // tile_dim1 @ [143:128]
  g1[5] = (int)stride0;                            // tensor_dim0_stride[31:0]
  g1[6] = 0;
  g1[7] = 0;
  const tdm_i32x4 z4 = (tdm_i32x4){0, 0, 0, 0};
#if __clang_major__ >= 23
  const tdm_i32x8 z8 = (tdm_i32x8){0, 0, 0, 0, 0, 0, 0, 0};
  __builtin_amdgcn_tensor_load_to_lds(g0, g1, z4, z4, z8, 0);
#else
  __builtin_amdgcn_tensor_load_to_lds(g0, g1, z4, z4, 0);
#endif
}
#endif  // USE_TDM

// -------------------------- WMMA fragment helpers --------------------------
// 16-bit A-matrix 16x32 (ISA 7.12.2): lane(l<16): row M=l, K in {0..7,16..23};
// lane(l>=16): row M=l-16, K in {8..15,24..31}; 2 bf16 per VGPR.
__device__ __forceinline__ v16bf load_a_frag(const __bf16* __restrict__ base, int ld) {
  const int lane = threadIdx.x & 31;
  const int row  = lane & 15;
  const int hi   = lane >> 4;
  const __bf16* rp = base + row * ld;
  v16bf a;
#pragma unroll
  for (int i = 0; i < 8; ++i) {
    const int kk = ((i & 4) ? 16 : 0) + hi * 8 + (i & 3) * 2;
    a[2 * i]     = rp[kk];
    a[2 * i + 1] = rp[kk + 1];
  }
  return a;
}

// 16-bit B-matrix 32x16 read from a K-major ("transposed") buffer where
// element (kk, col) lives at base[col*ldT + kk].
__device__ __forceinline__ v16bf load_b_fragT(const __bf16* __restrict__ base, int ldT) {
  const int lane = threadIdx.x & 31;
  const int col  = lane & 15;
  const int hi   = lane >> 4;
  const __bf16* cp = base + col * ldT;
  v16bf b;
#pragma unroll
  for (int j = 0; j < 8; ++j) {
    const int kk = ((j & 4) ? 16 : 0) + hi * 8 + (j & 3) * 2;
    b[2 * j]     = cp[kk];
    b[2 * j + 1] = cp[kk + 1];
  }
  return b;
}

__device__ __forceinline__ v8f wmma_bf16(v16bf a, v16bf b, v8f c) {
  return __builtin_amdgcn_wmma_f32_16x16x32_bf16(
      /*neg_a=*/false, a, /*neg_b=*/false, b,
      /*c_mod=*/(short)0, c, /*reuse_a=*/false, /*reuse_b=*/false);
}

// ----------------------------- fp32 -> bf16 --------------------------------
__global__ void mha_cvt_bf16(const float* __restrict__ in, __bf16* __restrict__ out, int n) {
  int i = blockIdx.x * blockDim.x + threadIdx.x;
  const int stride = gridDim.x * blockDim.x;
  for (; i < n; i += stride) out[i] = (__bf16)in[i];
}

// --------------------------- 128x128 bf16 GEMM -----------------------------
// C[M,N] = A[M,K] * B[K,N].  headSplit!=0 : write bf16 to [B,H,S,HD] layout.
// headSplit==0 : write fp32 (+bias) row-major.
__global__ __launch_bounds__(256) void mha_gemm_bf16(
    const __bf16* __restrict__ A, const __bf16* __restrict__ Bw,
    int M, int N, int K,
    float* outF, const float* __restrict__ bias,
    __bf16* outB, int headSplit) {
  __shared__ __bf16 sA[128 * 32];   // [row][k]   row-major
  __shared__ __bf16 sBt[128 * 32];  // [col][k]   K-major (transposed)

  const int t    = threadIdx.x;
  const int lane = t & 31;
  const int wave = t >> 5;
  const int hi   = lane >> 4;
  const int l16  = lane & 15;
  const int m0   = blockIdx.y * 128;
  const int n0   = blockIdx.x * 128;
  const int mb   = (wave & 3) * 32;   // wave's m sub-tile base (2 x 16)
  const int nb   = (wave >> 2) * 64;  // wave's n sub-tile base (4 x 16)

  v8f c[2][4];
#pragma unroll
  for (int i = 0; i < 2; ++i)
#pragma unroll
    for (int j = 0; j < 4; ++j) c[i][j] = (v8f){};

  for (int k0 = 0; k0 < K; k0 += 32) {
    __syncthreads();
#if USE_TDM
    // Wave 0 DMAs the whole 128x32 A tile into LDS via the Tensor Data Mover.
    if (wave == 0) {
      tdm_load_2d_bf16((unsigned)(size_t)(void*)sA,
                       A + (size_t)m0 * K + k0,
                       /*tensor_d0=*/32, /*tensor_d1=*/128,
                       /*tile_d0=*/32, /*tile_d1=*/128, /*stride0=*/(unsigned)K);
    }
#else
    {  // A tile: 256 threads x 16 bf16 (half a row each)
      const int row = t >> 1, half = t & 1;
      const uint4* src = (const uint4*)(A + (size_t)(m0 + row) * K + k0 + half * 16);
      uint4* dst = (uint4*)(sA + row * 32 + half * 16);
      dst[0] = src[0];
      dst[1] = src[1];
    }
#endif
    {  // B tile, stored transposed: thread handles (k=t>>3, 16 cols)
      const int kr = t >> 3, seg = t & 7;
      const __bf16* src = Bw + (size_t)(k0 + kr) * N + n0 + seg * 16;
#pragma unroll
      for (int i = 0; i < 16; ++i) sBt[(seg * 16 + i) * 32 + kr] = src[i];
      if (k0 + 32 < K)  // prefetch next B tile rows into cache
        __builtin_prefetch(Bw + (size_t)(k0 + 32 + kr) * N + n0 + seg * 16, 0, 3);
    }
#if USE_TDM
    if (wave == 0) __builtin_amdgcn_s_wait_tensorcnt(0);
#endif
    __syncthreads();

    const v16bf a0 = load_a_frag(sA + (mb + 0) * 32, 32);
    const v16bf a1 = load_a_frag(sA + (mb + 16) * 32, 32);
#pragma unroll
    for (int j = 0; j < 4; ++j) {
      const v16bf b = load_b_fragT(sBt + (nb + j * 16) * 32, 32);
      c[0][j] = wmma_bf16(a0, b, c[0][j]);
      c[1][j] = wmma_bf16(a1, b, c[1][j]);
    }
  }

  // Epilogue. C layout: lane group g=hi holds rows M=r+8g, col N=l16.
#pragma unroll
  for (int i = 0; i < 2; ++i)
#pragma unroll
    for (int j = 0; j < 4; ++j)
#pragma unroll
      for (int r = 0; r < 8; ++r) {
        const int row = m0 + mb + i * 16 + r + hi * 8;
        const int col = n0 + nb + j * 16 + l16;
        const float v = c[i][j][r];
        if (headSplit) {
          const int bb = row >> 11, s = row & (S_LEN - 1);
          const int h = col >> 6, d = col & (HDIM - 1);
          outB[(((size_t)bb * NHEAD + h) * S_LEN + s) * HDIM + d] = (__bf16)v;
        } else {
          outF[(size_t)row * N + col] = v + (bias ? bias[col] : 0.0f);
        }
      }
}

// ------------------------- flash attention (causal) ------------------------
// One wave per 16-query tile; online softmax; Q*K^T and P*V via WMMA.
// K tile staged by per-wave TDM (TENSORcnt is per-wave, matching per-wave
// private tiles); V staged manually because it must land transposed.
__global__ __launch_bounds__(AW * 32) void mha_flash_attn(
    const __bf16* __restrict__ Q, const __bf16* __restrict__ K,
    const __bf16* __restrict__ V, __bf16* __restrict__ ctx) {
  __shared__ __bf16 lds[AW * 4608];  // per wave: K 32x64, V^T 64x32, P 16x32

  const int lane = threadIdx.x & 31;
  const int wave = threadIdx.x >> 5;
  const int hi   = lane >> 4;
  const int l16  = lane & 15;

  __bf16* ldsK  = lds + wave * 4608;  // [key][d]   32 x 64
  __bf16* ldsVt = ldsK + 2048;        // [d][key]   64 x 32 (K-major for B-frag)
  __bf16* ldsP  = ldsVt + 2048;       // [q][key]   16 x 32

  const int task = blockIdx.x * AW + wave;  // B*H*(S/16) = 8192 tasks
  const int qi   = task & (S_LEN / 16 - 1);
  const int bh   = task >> 7;
  const int q0   = qi * 16;

  const __bf16* Qbh = Q + (size_t)bh * S_LEN * HDIM;
  const __bf16* Kbh = K + (size_t)bh * S_LEN * HDIM;
  const __bf16* Vbh = V + (size_t)bh * S_LEN * HDIM;

  const v16bf qa0 = load_a_frag(Qbh + q0 * HDIM + 0, HDIM);
  const v16bf qa1 = load_a_frag(Qbh + q0 * HDIM + 32, HDIM);

  v8f acc[4];
  float mrun[8], lrun[8];
#pragma unroll
  for (int j = 0; j < 4; ++j) acc[j] = (v8f){};
#pragma unroll
  for (int r = 0; r < 8; ++r) { mrun[r] = -3.0e38f; lrun[r] = 0.0f; }

  const int kend = q0 + 16;          // causal: keys [0, kend)
  const int nkb  = (kend + 31) >> 5; // 32-key blocks

  for (int kb = 0; kb < nkb; ++kb) {
    const int k0 = kb * 32;
#if USE_TDM
    // DMA the 32x64 K tile into this wave's LDS region.
    tdm_load_2d_bf16((unsigned)(size_t)(void*)ldsK,
                     Kbh + (size_t)k0 * HDIM,
                     /*tensor_d0=*/HDIM, /*tensor_d1=*/(unsigned)(S_LEN - k0),
                     /*tile_d0=*/HDIM, /*tile_d1=*/32, /*stride0=*/HDIM);
#else
    {  // stage K block rows (each lane copies one 64-elem key row)
      const uint4* src = (const uint4*)(Kbh + (size_t)(k0 + lane) * HDIM);
      uint4* dst = (uint4*)(ldsK + lane * HDIM);
#pragma unroll
      for (int i = 0; i < 8; ++i) dst[i] = src[i];
    }
#endif
    {  // stage V block transposed ([d][key])
      uint4 vr[8];
      const uint4* src = (const uint4*)(Vbh + (size_t)(k0 + lane) * HDIM);
#pragma unroll
      for (int i = 0; i < 8; ++i) vr[i] = src[i];
      const __bf16* vb = (const __bf16*)vr;
#pragma unroll
      for (int d = 0; d < HDIM; ++d) ldsVt[d * 32 + lane] = vb[d];
    }
#if USE_TDM
    __builtin_amdgcn_s_wait_tensorcnt(0);
#endif
    asm volatile("s_wait_dscnt 0" ::: "memory");  // cross-lane LDS visibility

    // scores S = Q * K^T  (two 16x16 key tiles, HD=64 => 2 WMMA each)
    v8f sc0 = (v8f){}, sc1 = (v8f){};
    sc0 = wmma_bf16(qa0, load_b_fragT(ldsK + 0 * HDIM + 0, HDIM), sc0);
    sc0 = wmma_bf16(qa1, load_b_fragT(ldsK + 0 * HDIM + 32, HDIM), sc0);
    sc1 = wmma_bf16(qa0, load_b_fragT(ldsK + 16 * HDIM + 0, HDIM), sc1);
    sc1 = wmma_bf16(qa1, load_b_fragT(ldsK + 16 * HDIM + 32, HDIM), sc1);

    float fac[8];
#pragma unroll
    for (int r = 0; r < 8; ++r) {
      const int qrow = q0 + r + hi * 8;
      float v0 = sc0[r] * SCALE;
      float v1 = sc1[r] * SCALE;
      if (k0 + l16 > qrow)      v0 = -3.0e38f;   // causal mask
      if (k0 + 16 + l16 > qrow) v1 = -3.0e38f;
      // row max across the 16 lanes of this half-wave (N=0..15)
      float mx = fmaxf(v0, v1);
      mx = fmaxf(mx, __shfl_xor(mx, 1, 32));
      mx = fmaxf(mx, __shfl_xor(mx, 2, 32));
      mx = fmaxf(mx, __shfl_xor(mx, 4, 32));
      mx = fmaxf(mx, __shfl_xor(mx, 8, 32));
      const float mn = fmaxf(mrun[r], mx);
      fac[r] = exp2f((mrun[r] - mn) * LOG2E);
      const float p0 = exp2f((v0 - mn) * LOG2E);
      const float p1 = exp2f((v1 - mn) * LOG2E);
      float ps = p0 + p1;
      ps += __shfl_xor(ps, 1, 32);
      ps += __shfl_xor(ps, 2, 32);
      ps += __shfl_xor(ps, 4, 32);
      ps += __shfl_xor(ps, 8, 32);
      mrun[r] = mn;
      lrun[r] = lrun[r] * fac[r] + ps;
      // C-layout -> row-major P in LDS (for A-fragment reload)
      ldsP[(r + hi * 8) * 32 + l16]      = (__bf16)p0;
      ldsP[(r + hi * 8) * 32 + 16 + l16] = (__bf16)p1;
    }
#pragma unroll
    for (int nt = 0; nt < 4; ++nt)
#pragma unroll
      for (int r = 0; r < 8; ++r) acc[nt][r] *= fac[r];

    asm volatile("s_wait_dscnt 0" ::: "memory");
    const v16bf pa = load_a_frag(ldsP, 32);
#pragma unroll
    for (int nt = 0; nt < 4; ++nt)
      acc[nt] = wmma_bf16(pa, load_b_fragT(ldsVt + nt * 16 * 32, 32), acc[nt]);
  }

  // normalize + write ctx in [B, S, H*HD] (= bqhd reshape) as bf16
#pragma unroll
  for (int r = 0; r < 8; ++r) lrun[r] = 1.0f / lrun[r];
  const int bb = bh >> 4, h = bh & (NHEAD - 1);
#pragma unroll
  for (int nt = 0; nt < 4; ++nt)
#pragma unroll
    for (int r = 0; r < 8; ++r) {
      const int row = q0 + r + hi * 8;
      const int col = h * HDIM + nt * 16 + l16;
      ctx[((size_t)bb * S_LEN + row) * DMODEL + col] = (__bf16)(acc[nt][r] * lrun[r]);
    }
}

// ------------------------------- launcher ----------------------------------
extern "C" void kernel_launch(void* const* d_in, const int* in_sizes, int n_in,
                              void* d_out, int out_size, void* d_ws, size_t ws_size,
                              hipStream_t stream) {
  (void)in_sizes; (void)n_in; (void)out_size; (void)ws_size;
  const float* x  = (const float*)d_in[0];
  const float* Wq = (const float*)d_in[1];
  const float* Wk = (const float*)d_in[2];
  const float* Wv = (const float*)d_in[3];
  const float* Wo = (const float*)d_in[4];
  const float* bo = (const float*)d_in[5];
  float* out = (float*)d_out;

  const size_t MSZ = (size_t)MROWS * DMODEL;   // 8,388,608
  const size_t WSZ = (size_t)DMODEL * DMODEL;  // 1,048,576
  __bf16* xb   = (__bf16*)d_ws;
  __bf16* wqb  = xb + MSZ;
  __bf16* wkb  = wqb + WSZ;
  __bf16* wvb  = wkb + WSZ;
  __bf16* wob  = wvb + WSZ;
  __bf16* Qb   = wob + WSZ;
  __bf16* Kb   = Qb + MSZ;
  __bf16* Vb   = Kb + MSZ;
  __bf16* ctxb = Vb + MSZ;  // total = 5*MSZ + 4*WSZ bf16 ~ 92.3 MB

  mha_cvt_bf16<<<2048, 256, 0, stream>>>(x, xb, (int)MSZ);
  mha_cvt_bf16<<<1024, 256, 0, stream>>>(Wq, wqb, (int)WSZ);
  mha_cvt_bf16<<<1024, 256, 0, stream>>>(Wk, wkb, (int)WSZ);
  mha_cvt_bf16<<<1024, 256, 0, stream>>>(Wv, wvb, (int)WSZ);
  mha_cvt_bf16<<<1024, 256, 0, stream>>>(Wo, wob, (int)WSZ);

  const dim3 gg(DMODEL / 128, MROWS / 128);  // (8, 64)
  mha_gemm_bf16<<<gg, 256, 0, stream>>>(xb, wqb, MROWS, DMODEL, DMODEL,
                                        nullptr, nullptr, Qb, 1);
  mha_gemm_bf16<<<gg, 256, 0, stream>>>(xb, wkb, MROWS, DMODEL, DMODEL,
                                        nullptr, nullptr, Kb, 1);
  mha_gemm_bf16<<<gg, 256, 0, stream>>>(xb, wvb, MROWS, DMODEL, DMODEL,
                                        nullptr, nullptr, Vb, 1);

  mha_flash_attn<<<(4 * NHEAD * (S_LEN / 16)) / AW, AW * 32, 0, stream>>>(Qb, Kb, Vb, ctxb);

  mha_gemm_bf16<<<gg, 256, 0, stream>>>(ctxb, wob, MROWS, DMODEL, DMODEL,
                                        out, bo, nullptr, 0);
}
